// SigmoidAttention_2001454760023
// MI455X (gfx1250) — compile-verified, matrix operand order checked by reading, not run on hardware
//
#include <hip/hip_runtime.h>

#define TLEN 1024
#define BSZQ 4
#define EMB  1024
#define NHEAD 16
#define HDIM 64
#define NBH  (BSZQ * NHEAD)   // 64 heads total
#define MROWS (TLEN * BSZQ)   // 4096 rows of [T,B,E] flattened

typedef __bf16 bf16;
typedef __attribute__((ext_vector_type(16))) __bf16       v16bf;
typedef __attribute__((ext_vector_type(4)))  __bf16       v4bf;
typedef __attribute__((ext_vector_type(8)))  float        v8f;
typedef __attribute__((ext_vector_type(4)))  unsigned int u32x4;
typedef __attribute__((ext_vector_type(4)))  int          i32x4;

typedef __attribute__((address_space(1))) i32x4 gas_i32x4;
typedef __attribute__((address_space(3))) i32x4 las_i32x4;

union FragU {
  v16bf bf;
  u32x4 q[2];
};

__device__ __forceinline__ v8f wmma_bf16(v16bf a, v16bf b, v8f c) {
  // 8 args: (neg_a, A, neg_b, B, c_mod, C, reuse_a, reuse_b)
  return __builtin_amdgcn_wmma_f32_16x16x32_bf16(false, a, false, b, (short)0, c,
                                                 false, false);
}

__device__ __forceinline__ v8f v8f_zero() {
  v8f z = {0.0f, 0.0f, 0.0f, 0.0f, 0.0f, 0.0f, 0.0f, 0.0f};
  return z;
}

// ---- CDNA5 async global->LDS copy (ASYNCcnt path), with sync fallback ------
__device__ __forceinline__ void async_cp_b128(void* lds_dst, const void* gsrc) {
#if __has_builtin(__builtin_amdgcn_global_load_async_to_lds_b128)
  __builtin_amdgcn_global_load_async_to_lds_b128(
      (gas_i32x4*)gsrc, (las_i32x4*)lds_dst, 0, 0);
#else
  *(u32x4*)lds_dst = *(const u32x4*)gsrc;
#endif
}

__device__ __forceinline__ void wait_async0() {
#if __has_builtin(__builtin_amdgcn_s_wait_asynccnt)
  __builtin_amdgcn_s_wait_asynccnt(0);
#else
  asm volatile("s_wait_asynccnt 0" ::: "memory");
#endif
}

// 16x32 bf16 fragment from a row-major bf16 matrix (global or LDS pointer).
// Row index (M for A-operand, N for B-operand) = rbase + (lane&15).
// Lanes 0-15 carry k {0..7, 16..23}, lanes 16-31 carry k {8..15, 24..31}.
__device__ __forceinline__ v16bf load_frag_bf16(const bf16* base, int ld,
                                                int rbase, int k0, int lane) {
  const bf16* p =
      base + (size_t)(rbase + (lane & 15)) * ld + k0 + ((lane & 16) ? 8 : 0);
  FragU f;
  f.q[0] = *(const u32x4*)(p);
  f.q[1] = *(const u32x4*)(p + 16);
  return f.bf;
}

// ---------------------------------------------------------------------------
// Kernel 0: bulk f32 -> bf16 conversion (one pass; native v_cvt path)
// ---------------------------------------------------------------------------
__global__ __launch_bounds__(256) void cvt_f32_bf16_kernel(
    const float* __restrict__ src, bf16* __restrict__ dst, int n) {
  int i = (blockIdx.x * 256 + threadIdx.x) * 4;
  if (i + 3 < n) {
    const float4 v = *(const float4*)(src + i);
    v4bf o;
    o[0] = (bf16)v.x;
    o[1] = (bf16)v.y;
    o[2] = (bf16)v.z;
    o[3] = (bf16)v.w;
    *(v4bf*)(dst + i) = o;
  }
}

// ---------------------------------------------------------------------------
// Kernel 1: fused QKV projection.  grid.z = 0/1/2 selects q/k/v.
// out = x[4096,1024] @ W[1024,1024]^T + bias, stored bf16:
//   q -> qbuf[bh][t][d]  (scaled by 1/sqrt(EMB) = 1/32)
//   k -> kbuf[bh][s][d]
//   v -> vtbuf[bh][d][s]   (transposed for the P@V B-operand)
// ---------------------------------------------------------------------------
__global__ __launch_bounds__(256) void qkv_proj_kernel(
    const bf16* __restrict__ xq, const bf16* __restrict__ xk,
    const bf16* __restrict__ xv, const bf16* __restrict__ wqkv,
    const float* __restrict__ bqkv, bf16* __restrict__ qbuf,
    bf16* __restrict__ kbuf, bf16* __restrict__ vtbuf) {
  const int mode = blockIdx.z;
  const bf16* __restrict__ x = (mode == 0) ? xq : (mode == 1) ? xk : xv;
  const bf16* __restrict__ w = wqkv + (size_t)mode * EMB * EMB;
  const float* __restrict__ bias = bqkv + mode * EMB;

  const int lane = threadIdx.x & 31;
  const int wave = threadIdx.x >> 5;
  const int mbase = blockIdx.y * 128 + (wave & 3) * 32;  // 4 waves along M
  const int nbase = blockIdx.x * 64 + (wave >> 2) * 32;  // 2 waves along N

  v8f acc[2][2];
#pragma unroll
  for (int i = 0; i < 2; ++i)
#pragma unroll
    for (int j = 0; j < 2; ++j) acc[i][j] = v8f_zero();

  for (int k0 = 0; k0 < EMB; k0 += 32) {
    v16bf a0 = load_frag_bf16(x, EMB, mbase, k0, lane);
    v16bf a1 = load_frag_bf16(x, EMB, mbase + 16, k0, lane);
    v16bf b0 = load_frag_bf16(w, EMB, nbase, k0, lane);
    v16bf b1 = load_frag_bf16(w, EMB, nbase + 16, k0, lane);
    acc[0][0] = wmma_bf16(a0, b0, acc[0][0]);
    acc[0][1] = wmma_bf16(a0, b1, acc[0][1]);
    acc[1][0] = wmma_bf16(a1, b0, acc[1][0]);
    acc[1][1] = wmma_bf16(a1, b1, acc[1][1]);
  }

  const int mofs = (lane & 16) ? 8 : 0;
  const int nlo = lane & 15;
#pragma unroll
  for (int i = 0; i < 2; ++i)
#pragma unroll
    for (int j = 0; j < 2; ++j)
#pragma unroll
      for (int r = 0; r < 8; ++r) {
        int m = mbase + i * 16 + r + mofs;  // row of [T,B,E] flat
        int n = nbase + j * 16 + nlo;       // column within E
        float v = acc[i][j][r] + bias[n];
        int t = m >> 2, bb = m & 3;  // m = t*BSZ + b
        int h = n >> 6, d = n & 63;  // n = h*HDIM + d
        int bh = bb * NHEAD + h;
        if (mode == 0) {
          qbuf[((size_t)bh * TLEN + t) * HDIM + d] = (bf16)(v * 0.03125f);
        } else if (mode == 1) {
          kbuf[((size_t)bh * TLEN + t) * HDIM + d] = (bf16)v;
        } else {
          vtbuf[((size_t)bh * HDIM + d) * TLEN + t] = (bf16)v;
        }
      }
}

// ---------------------------------------------------------------------------
// Kernel 2: fused sigmoid attention.
// Per (head, 128 T rows); each wave owns 16 T rows, Q strip held in registers.
// K/V^T blocks are staged once per workgroup through LDS via async-to-LDS
// (all 8 waves share them).  Stream S in blocks of 32:
//   P = sigmoid(Q @ K^T) -> write attn (f32, once),
//   transpose P (C-layout -> A-layout) through per-wave LDS, O += P @ V.
// ---------------------------------------------------------------------------
__global__ __launch_bounds__(256) void sig_attn_kernel(
    const bf16* __restrict__ qbuf, const bf16* __restrict__ kbuf,
    const bf16* __restrict__ vtbuf, float* __restrict__ attn_out,
    bf16* __restrict__ obuf) {
  __shared__ bf16 klds[32][64];     // K rows s0..s0+31, 4 KB
  __shared__ bf16 vlds[64][32];     // V^T rows d=0..63, s-chunk, 4 KB
  __shared__ bf16 plds[8][16][32];  // per-wave P transpose region, 8 KB

  const int tid = threadIdx.x;
  const int lane = tid & 31;
  const int wave = tid >> 5;
  const int bh = blockIdx.y;
  const int t0 = blockIdx.x * 128 + wave * 16;

  const bf16* qh = qbuf + (size_t)bh * TLEN * HDIM;
  const bf16* kh = kbuf + (size_t)bh * TLEN * HDIM;
  const bf16* vh = vtbuf + (size_t)bh * HDIM * TLEN;
  float* ah = attn_out + (size_t)bh * TLEN * TLEN;

  // Q strip 16x64 resident in registers (2 K-chunks of 32)
  v16bf aq0 = load_frag_bf16(qh, HDIM, t0, 0, lane);
  v16bf aq1 = load_frag_bf16(qh, HDIM, t0, 32, lane);

  v8f o[4];
#pragma unroll
  for (int i = 0; i < 4; ++i) o[i] = v8f_zero();

  const int mofs = (lane & 16) ? 8 : 0;
  const int nlo = lane & 15;

  // staging assignments (256 threads, one b128 each per buffer)
  const int krow = tid >> 3, kch = tid & 7;  // 32 rows x 8 x 16B = 4 KB
  const int vrow = tid >> 2, vch = tid & 3;  // 64 rows x 4 x 16B = 4 KB

  for (int s0 = 0; s0 < TLEN; s0 += 32) {
    // ---- stage shared K / V^T blocks into LDS (ASYNCcnt path) ----
    async_cp_b128(&klds[krow][kch * 8], kh + (size_t)(s0 + krow) * HDIM + kch * 8);
    async_cp_b128(&vlds[vrow][vch * 8], vh + (size_t)vrow * TLEN + s0 + vch * 8);
    wait_async0();
    __syncthreads();

#pragma unroll
    for (int ss = 0; ss < 2; ++ss) {
      // B-operand of Q@K^T from LDS: lane n holds K-row (ss*16+n)
      v16bf kb0 = load_frag_bf16(&klds[0][0], HDIM, ss * 16, 0, lane);
      v16bf kb1 = load_frag_bf16(&klds[0][0], HDIM, ss * 16, 32, lane);
      v8f p = v8f_zero();
      p = wmma_bf16(aq0, kb0, p);
      p = wmma_bf16(aq1, kb1, p);
#pragma unroll
      for (int r = 0; r < 8; ++r) {
        float sg = 1.0f / (1.0f + __expf(-p[r]));
        // write attn[bh][t][s] exactly once (dominant HBM traffic)
        ah[(size_t)(t0 + r + mofs) * TLEN + (s0 + ss * 16 + nlo)] = sg;
        // stage bf16 P row-major in this wave's LDS region for the transpose
        plds[wave][r + mofs][ss * 16 + nlo] = (bf16)sg;
      }
    }
    // Re-read P as a 16x32 A-operand fragment (LDS transpose, wave-private)
    v16bf ap = load_frag_bf16(&plds[wave][0][0], 32, 0, 0, lane);
#pragma unroll
    for (int dsub = 0; dsub < 4; ++dsub) {
      // B-operand of P@V from LDS: lane n holds V^T row d = dsub*16+n
      v16bf vb = load_frag_bf16(&vlds[0][0], 32, dsub * 16, 0, lane);
      o[dsub] = wmma_bf16(ap, vb, o[dsub]);
    }
    __syncthreads();  // protect K/V staging buffers before next iteration
  }

  // Store O to obuf[T*B, E] bf16 for the output projection
  const int bb = bh >> 4, h = bh & 15;
#pragma unroll
  for (int dsub = 0; dsub < 4; ++dsub)
#pragma unroll
    for (int r = 0; r < 8; ++r) {
      int t = t0 + r + mofs;
      int m = t * BSZQ + bb;
      int n = h * HDIM + dsub * 16 + nlo;
      obuf[(size_t)m * EMB + n] = (bf16)o[dsub][r];
    }
}

// ---------------------------------------------------------------------------
// Kernel 3: output projection: out = O[4096,1024] @ Wout^T + bias  (f32 out)
// ---------------------------------------------------------------------------
__global__ __launch_bounds__(256) void out_proj_kernel(
    const bf16* __restrict__ obuf, const bf16* __restrict__ wout,
    const float* __restrict__ bout, float* __restrict__ out) {
  const int lane = threadIdx.x & 31;
  const int wave = threadIdx.x >> 5;
  const int mbase = blockIdx.y * 128 + (wave & 3) * 32;
  const int nbase = blockIdx.x * 64 + (wave >> 2) * 32;

  v8f acc[2][2];
#pragma unroll
  for (int i = 0; i < 2; ++i)
#pragma unroll
    for (int j = 0; j < 2; ++j) acc[i][j] = v8f_zero();

  for (int k0 = 0; k0 < EMB; k0 += 32) {
    v16bf a0 = load_frag_bf16(obuf, EMB, mbase, k0, lane);
    v16bf a1 = load_frag_bf16(obuf, EMB, mbase + 16, k0, lane);
    v16bf b0 = load_frag_bf16(wout, EMB, nbase, k0, lane);
    v16bf b1 = load_frag_bf16(wout, EMB, nbase + 16, k0, lane);
    acc[0][0] = wmma_bf16(a0, b0, acc[0][0]);
    acc[0][1] = wmma_bf16(a0, b1, acc[0][1]);
    acc[1][0] = wmma_bf16(a1, b0, acc[1][0]);
    acc[1][1] = wmma_bf16(a1, b1, acc[1][1]);
  }

  const int mofs = (lane & 16) ? 8 : 0;
  const int nlo = lane & 15;
#pragma unroll
  for (int i = 0; i < 2; ++i)
#pragma unroll
    for (int j = 0; j < 2; ++j)
#pragma unroll
      for (int r = 0; r < 8; ++r) {
        int m = mbase + i * 16 + r + mofs;
        int n = nbase + j * 16 + nlo;
        out[(size_t)m * EMB + n] = acc[i][j][r] + bout[n];
      }
}

// ---------------------------------------------------------------------------
extern "C" void kernel_launch(void* const* d_in, const int* in_sizes, int n_in,
                              void* d_out, int out_size, void* d_ws, size_t ws_size,
                              hipStream_t stream) {
  const float* qin  = (const float*)d_in[0];
  const float* kin  = (const float*)d_in[1];
  const float* vin  = (const float*)d_in[2];
  const float* wqkv = (const float*)d_in[3];
  const float* bqkv = (const float*)d_in[4];
  const float* wout = (const float*)d_in[5];
  const float* bout = (const float*)d_in[6];

  float* out_attn_output = (float*)d_out;                          // [T,B,E]
  float* out_attn        = out_attn_output + (size_t)MROWS * EMB;  // [BH,T,S]

  // Workspace layout (bf16 elements):
  //   xq|xk|xv (3 x 4M) | wqkv (3M) | wout (1M) | q|k|vT (3 x 4M) | O (4M)
  bf16* xqb   = (bf16*)d_ws;
  bf16* xkb   = xqb + (size_t)MROWS * EMB;
  bf16* xvb   = xkb + (size_t)MROWS * EMB;
  bf16* wqkvb = xvb + (size_t)MROWS * EMB;
  bf16* woutb = wqkvb + (size_t)3 * EMB * EMB;
  bf16* qbuf  = woutb + (size_t)EMB * EMB;
  bf16* kbuf  = qbuf + (size_t)NBH * TLEN * HDIM;
  bf16* vtbuf = kbuf + (size_t)NBH * TLEN * HDIM;
  bf16* obuf  = vtbuf + (size_t)NBH * TLEN * HDIM;

  dim3 blk(256);
  const int NX = MROWS * EMB;  // 4 M elements
  cvt_f32_bf16_kernel<<<NX / 1024, blk, 0, stream>>>(qin, xqb, NX);
  cvt_f32_bf16_kernel<<<NX / 1024, blk, 0, stream>>>(kin, xkb, NX);
  cvt_f32_bf16_kernel<<<NX / 1024, blk, 0, stream>>>(vin, xvb, NX);
  cvt_f32_bf16_kernel<<<(3 * EMB * EMB) / 1024, blk, 0, stream>>>(wqkv, wqkvb,
                                                                  3 * EMB * EMB);
  cvt_f32_bf16_kernel<<<(EMB * EMB) / 1024, blk, 0, stream>>>(wout, woutb,
                                                              EMB * EMB);

  qkv_proj_kernel<<<dim3(EMB / 64, MROWS / 128, 3), blk, 0, stream>>>(
      xqb, xkb, xvb, wqkvb, bqkv, qbuf, kbuf, vtbuf);
  sig_attn_kernel<<<dim3(TLEN / 128, NBH), blk, 0, stream>>>(
      qbuf, kbuf, vtbuf, out_attn, obuf);
  out_proj_kernel<<<dim3(EMB / 64, MROWS / 128), blk, 0, stream>>>(
      obuf, woutb, bout, out_attn_output);
}
